// DuplicateRemovalLayer_70325794505465
// MI455X (gfx1250) — matrix operand order repaired and checked by reading, not compiled, not run on hardware
//
#include <hip/hip_runtime.h>
#include <hip/hip_bf16.h>

// ---------------------------------------------------------------------------
// CDNA5 (gfx1250) wave32 WMMA implementation of the DuplicateRemovalLayer.
// Geometry MLP collapsed exactly: (rel@Wg1+bg1)@Wg2+bg2 = rel@(Wg1@Wg2)+const.
// ---------------------------------------------------------------------------

typedef __attribute__((ext_vector_type(16))) __bf16 v16bf;
typedef __attribute__((ext_vector_type(8)))  float  v8f;
typedef __attribute__((ext_vector_type(4)))  unsigned int v4u;
typedef __attribute__((ext_vector_type(8)))  int v8i;
typedef __attribute__((ext_vector_type(4)))  int v4i;

constexpr int kB = 8;
constexpr int kN = 512;
constexpr int kC = 256;
constexpr int kU = 128;

// ---- workspace layout (bytes) ----
constexpr size_t OFF_ORDER = 0;                      // 4096 int
constexpr size_t OFF_GEOM  = 16384;                  // 6 arrays x 4096 f32
constexpr size_t OFF_WGE   = OFF_GEOM + 6 * 16384;   // 114688 : 5 f32 (pad 256)
constexpr size_t OFF_WRANK = OFF_WGE + 256;          // tiled-T bf16 [4][128][32]
constexpr size_t OFF_WFEAT = OFF_WRANK + 32768;      // tiled-T bf16 [8][128][32]
constexpr size_t OFF_WK    = OFF_WFEAT + 65536;
constexpr size_t OFF_WQ    = OFF_WK + 32768;
constexpr size_t OFF_WV    = OFF_WQ + 32768;
constexpr size_t OFF_KB    = OFF_WV + 32768;         // k rows bf16 [B][N][U]
constexpr size_t OFF_QB    = OFF_KB + 1048576;       // q rows bf16 [B][N][U]
constexpr size_t OFF_VT    = OFF_QB + 1048576;       // v transposed bf16 [B][U][N]

// ---------------------------------------------------------------------------
// WMMA helpers (ISA 7.12.2 layouts, wave32)
// ---------------------------------------------------------------------------
__device__ __forceinline__ v8f wmma_bf16(v16bf a, v16bf b, v8f c) {
  return __builtin_amdgcn_wmma_f32_16x16x32_bf16(false, a, false, b, (short)0, c,
                                                 false, false);
}

// A fragment: 16x32 bf16 tile from row-major source (leading dim ld, elements).
// Lane L(0..15): row M=L, K {0..7,16..23}; lane L+16: same row, K {8..15,24..31}.
// Two contiguous 16B runs -> 2x ds_load_b128.
__device__ __forceinline__ v16bf load_a_frag(const __bf16* src, int ld) {
  const int lane = threadIdx.x & 31;
  const __bf16* row = src + (lane & 15) * ld + ((lane >> 4) << 3);
  union { uint4 u[2]; v16bf v; } c;
  c.u[0] = *(const uint4*)(row);
  c.u[1] = *(const uint4*)(row + 16);
  return c.v;
}

__device__ __forceinline__ v16bf load_a_frag_f32(const float* src, int ld) {
  const int lane = threadIdx.x & 31;
  const float* row = src + (lane & 15) * ld + ((lane >> 4) << 3);
  float4 a0 = *(const float4*)(row);
  float4 a1 = *(const float4*)(row + 4);
  float4 a2 = *(const float4*)(row + 16);
  float4 a3 = *(const float4*)(row + 20);
  v16bf f;
  f[0] = (__bf16)a0.x;  f[1] = (__bf16)a0.y;  f[2] = (__bf16)a0.z;  f[3] = (__bf16)a0.w;
  f[4] = (__bf16)a1.x;  f[5] = (__bf16)a1.y;  f[6] = (__bf16)a1.z;  f[7] = (__bf16)a1.w;
  f[8] = (__bf16)a2.x;  f[9] = (__bf16)a2.y;  f[10] = (__bf16)a2.z; f[11] = (__bf16)a2.w;
  f[12] = (__bf16)a3.x; f[13] = (__bf16)a3.y; f[14] = (__bf16)a3.z; f[15] = (__bf16)a3.w;
  return f;
}

// B fragment from K-contiguous (transposed) source: B[k][n] = src[n*ld + k].
// Per lane K {0..7,16..23} (lane<16) / {8..15,24..31} -> 2x ds_load_b128.
__device__ __forceinline__ v16bf load_b_frag_T(const __bf16* src, int ld) {
  const int lane = threadIdx.x & 31;
  const __bf16* row = src + (lane & 15) * ld + ((lane >> 4) << 3);
  union { uint4 u[2]; v16bf v; } c;
  c.u[0] = *(const uint4*)(row);
  c.u[1] = *(const uint4*)(row + 16);
  return c.v;
}

// ---------------------------------------------------------------------------
// Tensor Data Mover: issue a 1D/2D tile load into LDS (D# per ISA 8.3/8.4).
// count=1, type=2("image"), data_size=1 (2-byte elements), no padding/iterate.
// clang-23 builtin form: (v4u g0, v8i g1, v4i g2, v4i g3, v8i g4, i32 cpol).
// ---------------------------------------------------------------------------
__device__ __forceinline__ void tdm_load_2d(unsigned lds_addr, const void* gptr,
                                            unsigned tensor_d0, unsigned tensor_d1,
                                            unsigned tile_d0, unsigned tile_d1,
                                            unsigned long long stride0) {
  const unsigned long long ga = (unsigned long long)(uintptr_t)gptr;
  v4u g0;
  g0[0] = 1u;                                            // count=1, user mode
  g0[1] = lds_addr;                                      // lds_addr
  g0[2] = (unsigned)(ga & 0xFFFFFFFFu);                  // global_addr[31:0]
  g0[3] = (unsigned)((ga >> 32) & 0x01FFFFFFu) | (2u << 30);  // addr hi | type=2
  v8i g1;
  g1[0] = (int)(1u << 16);                               // data_size=1 (2B)
  g1[1] = (int)((tensor_d0 & 0xFFFFu) << 16);            // tensor_dim0 lo
  g1[2] = (int)((tensor_d0 >> 16) | ((tensor_d1 & 0xFFFFu) << 16));
  g1[3] = (int)((tensor_d1 >> 16) | (tile_d0 << 16));    // tile_dim0
  g1[4] = (int)(tile_d1 & 0xFFFFu);                      // tile_dim1 (tile_dim2=0)
  g1[5] = (int)(stride0 & 0xFFFFFFFFull);                // tensor_dim0_stride lo
  g1[6] = (int)((stride0 >> 32) & 0xFFFFull);            // stride hi (stride1=0)
  g1[7] = 0;
  const v4i z4 = {0, 0, 0, 0};
  const v8i z8 = {0, 0, 0, 0, 0, 0, 0, 0};
  __builtin_amdgcn_tensor_load_to_lds(g0, g1, z4, z4, z8, 0);
}

// ---------------------------------------------------------------------------
// Kernel 0: weight f32->bf16 conversion into tiled-transposed layout
// dW[kt][c][kj] = W[kt*32+kj][c]  (kt = K/32 tiles) + geometry-MLP collapse.
// ---------------------------------------------------------------------------
__global__ void prep_convert_kernel(
    const float* __restrict__ Wrank, const float* __restrict__ Wfeat,
    const float* __restrict__ Wk, const float* __restrict__ Wq,
    const float* __restrict__ Wv,
    const float* __restrict__ Wg1, const float* __restrict__ bg1,
    const float* __restrict__ Wg2, const float* __restrict__ bg2,
    __bf16* __restrict__ dWrank, __bf16* __restrict__ dWfeat,
    __bf16* __restrict__ dWk, __bf16* __restrict__ dWq,
    __bf16* __restrict__ dWv, float* __restrict__ wge) {
  const int idx = blockIdx.x * blockDim.x + threadIdx.x;
  const float* src = nullptr;
  __bf16* dst = nullptr;
  int rel = 0;
  if (idx < 16384)       { src = Wrank; dst = dWrank; rel = idx; }
  else if (idx < 49152)  { src = Wfeat; dst = dWfeat; rel = idx - 16384; }
  else if (idx < 65536)  { src = Wk;    dst = dWk;    rel = idx - 49152; }
  else if (idx < 81920)  { src = Wq;    dst = dWq;    rel = idx - 65536; }
  else if (idx < 98304)  { src = Wv;    dst = dWv;    rel = idx - 81920; }
  if (src) {
    const int k = rel >> 7, c = rel & 127;
    const int kt = k >> 5, kj = k & 31;
    dst[(kt * 128 + c) * 32 + kj] = (__bf16)src[rel];
  }
  if (blockIdx.x == 0 && threadIdx.x < 5) {
    const int j = threadIdx.x;
    float acc = 0.f;
    if (j < 4) {
      for (int u = 0; u < kU; ++u) acc += Wg1[j * kU + u] * Wg2[u];
    } else {
      acc = bg2[0];
      for (int u = 0; u < kU; ++u) acc += bg1[u] * Wg2[u];
    }
    wge[j] = acc;
  }
}

// ---------------------------------------------------------------------------
// Kernel 1: per-box geometry precompute (cx, cy, w, h, log w, log h)
// ---------------------------------------------------------------------------
__global__ void prep_boxes_kernel(const float* __restrict__ boxes,
                                  float* __restrict__ gcx, float* __restrict__ gcy,
                                  float* __restrict__ gw, float* __restrict__ gh,
                                  float* __restrict__ glw, float* __restrict__ glh) {
  const int i = blockIdx.x * blockDim.x + threadIdx.x;
  if (i >= kB * kN) return;
  const float x0 = boxes[i * 4 + 0], y0 = boxes[i * 4 + 1];
  const float x1 = boxes[i * 4 + 2], y1 = boxes[i * 4 + 3];
  const float w = x1 - x0, h = y1 - y0;
  gcx[i] = 0.5f * (x0 + x1);
  gcy[i] = 0.5f * (y0 + y1);
  gw[i]  = w;
  gh[i]  = h;
  glw[i] = __logf(w);
  glh[i] = __logf(h);
}

// ---------------------------------------------------------------------------
// Kernel 2: descending argsort of scores (bitonic, LDS, one block per batch).
// 64-bit key = (score bits << 32) | ~index  -> stable, descending.
// ---------------------------------------------------------------------------
__global__ __launch_bounds__(512) void argsort_kernel(
    const float* __restrict__ scores, int* __restrict__ order) {
  __shared__ unsigned long long key[kN];
  const int tid = threadIdx.x;
  const int b = blockIdx.x;
  const float s = scores[b * kN + tid];
  key[tid] = (((unsigned long long)__float_as_uint(s)) << 32) |
             (unsigned long long)(0xFFFFFFFFu - (unsigned)tid);
  __syncthreads();
  for (int k = 2; k <= kN; k <<= 1) {
    for (int j = k >> 1; j > 0; j >>= 1) {
      const int ixj = tid ^ j;
      if (ixj > tid) {
        const unsigned long long a = key[tid], c = key[ixj];
        const bool desc = ((tid & k) == 0);
        if (desc ? (a < c) : (a > c)) { key[tid] = c; key[ixj] = a; }
      }
      __syncthreads();
    }
  }
  order[b * kN + tid] = (int)(0xFFFFFFFFu - (unsigned)(key[tid] & 0xFFFFFFFFull));
}

// ---------------------------------------------------------------------------
// Kernel 3: fused emb = enc@Wrank + feat@Wfeat (+bias), then k/q/v projections.
// One block = 16 rows, 4 waves x 32 output columns. bf16 WMMA, f32 accumulate.
// v is written transposed ([B][U][N]) with packed b128 stores.
// ---------------------------------------------------------------------------
__global__ __launch_bounds__(128) void emb_kqv_kernel(
    const float* __restrict__ feat, const float* __restrict__ enc_table,
    const int* __restrict__ order,
    const __bf16* __restrict__ Wrank, const __bf16* __restrict__ Wfeat,
    const float* __restrict__ b_rank, const float* __restrict__ b_feat,
    const __bf16* __restrict__ Wk, const __bf16* __restrict__ Wq,
    const __bf16* __restrict__ Wv,
    const float* __restrict__ bk, const float* __restrict__ bq,
    const float* __restrict__ bv,
    __bf16* __restrict__ kout, __bf16* __restrict__ qout,
    __bf16* __restrict__ vT) {
  alignas(16) __shared__ __bf16 sEnc[16][kU];    // gathered rank encodings
  alignas(16) __shared__ __bf16 sFeat[16][kC];   // appearance features
  alignas(16) __shared__ float  sEmb[16][kU];    // f32 embedding tile
  alignas(16) __shared__ __bf16 sWtT[kU][32];    // weight K-tile, K-contiguous

  const int tid  = threadIdx.x;
  const int wave = tid >> 5;
  const int lane = tid & 31;
  const int b    = blockIdx.x >> 5;
  const int n0   = (blockIdx.x & 31) << 4;
  const int rb   = b * kN + n0;

  for (int e = tid; e < 16 * kU; e += 128) {
    const int r = e >> 7, c = e & 127;
    sEnc[r][c] = (__bf16)enc_table[order[rb + r] * kU + c];
  }
  for (int e = tid; e < 16 * kC; e += 128) {
    const int r = e >> 8, c = e & 255;
    sFeat[r][c] = (__bf16)feat[(rb + r) * kC + c];
  }

  const int col0 = wave << 5;
  const int cn   = lane & 15;
  const int rsel = (lane >> 4) << 3;

  v8f acc0 = {}, acc1 = {};
  for (int kk = 0; kk < kU; kk += 32) {          // enc @ Wrank
    __syncthreads();
    for (int e = tid; e < 512; e += 128)         // tile kt = kk/32 (kk*kU elems in)
      ((uint4*)sWtT)[e] = ((const uint4*)(Wrank + kk * kU))[e];
    if (kk + 32 < kU)
      __builtin_prefetch(Wrank + (kk + 32) * kU + tid * 32, 0, 1);
    __syncthreads();
    v16bf a = load_a_frag(&sEnc[0][kk], kU);
    acc0 = wmma_bf16(a, load_b_frag_T(&sWtT[col0][0], 32), acc0);
    acc1 = wmma_bf16(a, load_b_frag_T(&sWtT[col0 + 16][0], 32), acc1);
  }
  for (int kk = 0; kk < kC; kk += 32) {          // feat @ Wfeat
    __syncthreads();
    for (int e = tid; e < 512; e += 128)
      ((uint4*)sWtT)[e] = ((const uint4*)(Wfeat + kk * kU))[e];
    if (kk + 32 < kC)
      __builtin_prefetch(Wfeat + (kk + 32) * kU + tid * 32, 0, 1);
    __syncthreads();
    v16bf a = load_a_frag(&sFeat[0][kk], kC);
    acc0 = wmma_bf16(a, load_b_frag_T(&sWtT[col0][0], 32), acc0);
    acc1 = wmma_bf16(a, load_b_frag_T(&sWtT[col0 + 16][0], 32), acc1);
  }
  const float bias0 = b_rank[col0 + cn] + b_feat[col0 + cn];
  const float bias1 = b_rank[col0 + 16 + cn] + b_feat[col0 + 16 + cn];
#pragma unroll
  for (int g = 0; g < 8; ++g) {
    sEmb[rsel + g][col0 + cn]      = acc0[g] + bias0;
    sEmb[rsel + g][col0 + 16 + cn] = acc1[g] + bias1;
  }

  const __bf16* Wp[3] = {Wk, Wq, Wv};
  const float*  bp[3] = {bk, bq, bv};
#pragma unroll 1
  for (int p = 0; p < 3; ++p) {
    v8f p0 = {}, p1 = {};
    for (int kk = 0; kk < kU; kk += 32) {
      __syncthreads();
      for (int e = tid; e < 512; e += 128)
        ((uint4*)sWtT)[e] = ((const uint4*)(Wp[p] + kk * kU))[e];
      __syncthreads();
      v16bf a = load_a_frag_f32(&sEmb[0][kk], kU);
      p0 = wmma_bf16(a, load_b_frag_T(&sWtT[col0][0], 32), p0);
      p1 = wmma_bf16(a, load_b_frag_T(&sWtT[col0 + 16][0], 32), p1);
    }
    const float pb0 = bp[p][col0 + cn];
    const float pb1 = bp[p][col0 + 16 + cn];
    if (p < 2) {                                  // k, q: row-major rows
      __bf16* op = (p == 0) ? kout : qout;
#pragma unroll
      for (int g = 0; g < 8; ++g) {
        op[(rb + rsel + g) * kU + col0 + cn]      = (__bf16)(p0[g] + pb0);
        op[(rb + rsel + g) * kU + col0 + 16 + cn] = (__bf16)(p1[g] + pb1);
      }
    } else {                                      // v: transposed, packed stores
      alignas(16) __bf16 t0[8], t1[8];
#pragma unroll
      for (int g = 0; g < 8; ++g) {
        t0[g] = (__bf16)(p0[g] + pb0);
        t1[g] = (__bf16)(p1[g] + pb1);
      }
      __bf16* d0p = vT + ((size_t)(b * kU + col0 + cn)) * kN + n0 + rsel;
      __bf16* d1p = vT + ((size_t)(b * kU + col0 + 16 + cn)) * kN + n0 + rsel;
      *(uint4*)d0p = *(const uint4*)t0;
      *(uint4*)d1p = *(const uint4*)t1;
    }
  }
}

// ---------------------------------------------------------------------------
// Kernel 4: fused relation attention (flash-attention style, no NxN tensors).
// Block = 16 query rows n; loop m in tiles of 64. q/vT tiles staged into LDS
// by the Tensor Data Mover (TENSORcnt), app subtiles per wave via WMMA,
// geometry+exp elementwise in f32, wts@v accumulated via WMMA over per-wave
// 32-column d slices. Z / logits reduced with ds_add_f32.
// ---------------------------------------------------------------------------
__global__ __launch_bounds__(128) void relation_kernel(
    const __bf16* __restrict__ kbuf, const __bf16* __restrict__ qbuf,
    const __bf16* __restrict__ vT,
    const float* __restrict__ gcx, const float* __restrict__ gcy,
    const float* __restrict__ gw, const float* __restrict__ gh,
    const float* __restrict__ glw, const float* __restrict__ glh,
    const float* __restrict__ wge,
    const float* __restrict__ Ws, const float* __restrict__ bs,
    const float* __restrict__ scores, float* __restrict__ out) {
  alignas(16) __shared__ __bf16 sK[16][kU];     // k rows (A of app GEMM)
  alignas(16) __shared__ __bf16 sQ[64][kU];     // q rows (B via transpose-read)
  alignas(16) __shared__ __bf16 sVT[kU][64];    // v columns, K(m)-contiguous
  alignas(16) __shared__ __bf16 sW[16][64];     // wts tile (A of second GEMM)
  __shared__ float sRow[6][16];
  __shared__ float sZ[16];
  __shared__ float sLogit[16];

  const int tid = threadIdx.x, wave = tid >> 5, lane = tid & 31;
  const int b  = blockIdx.x >> 5;
  const int n0 = (blockIdx.x & 31) << 4;
  const int rb = b * kN + n0;
  const int cn = lane & 15;
  const int rsel = (lane >> 4) << 3;
  const int d0 = wave << 5;

  if (tid < 16) { sZ[tid] = 0.f; sLogit[tid] = 0.f; }
  if (tid < 96) {
    const float* srcs[6] = {gcx, gcy, gw, gh, glw, glh};
    sRow[tid >> 4][tid & 15] = srcs[tid >> 4][rb + (tid & 15)];
  }
  for (int e = tid; e < 256; e += 128)
    ((uint4*)sK)[e] = ((const uint4*)(kbuf + rb * kU))[e];

  const float g0 = wge[0], g1 = wge[1], g2 = wge[2], g3 = wge[3], gB = wge[4];
  const float scale = 0.08838834764831845f;  // 1/sqrt(128)

  float z8[8] = {0.f, 0.f, 0.f, 0.f, 0.f, 0.f, 0.f, 0.f};
  v8f accd0 = {}, accd1 = {};

  for (int mt = 0; mt < kN; mt += 64) {
    __syncthreads();
    if (wave == 0) {
      // q tile: 64 rows x 128 cols, fully contiguous -> 1-D 8192-element tile
      tdm_load_2d((unsigned)(uintptr_t)&sQ[0][0],
                  qbuf + (size_t)(b * kN + mt) * kU,
                  (unsigned)(kN * kU), 1u, 8192u, 0u,
                  (unsigned long long)(kN * kU));
      // vT tile: 128 rows (d) x 64 cols (m), row stride kN elements
      tdm_load_2d((unsigned)(uintptr_t)&sVT[0][0],
                  vT + (size_t)b * kU * kN + mt,
                  (unsigned)kN, (unsigned)(kB * kU), 64u, 128u,
                  (unsigned long long)kN);
      __builtin_amdgcn_s_wait_tensorcnt(0);
    }
    __syncthreads();

    const int mloc = wave << 4;
    v8f app = {};
#pragma unroll
    for (int kk = 0; kk < kU; kk += 32) {
      v16bf a = load_a_frag(&sK[0][kk], kU);
      v16bf bfr = load_b_frag_T(&sQ[mloc][kk], kU);  // B[k][m] = q[m][k]
      app = wmma_bf16(a, bfr, app);
    }
    const int mcol = b * kN + mt + mloc + cn;
    const float cxm = gcx[mcol], cym = gcy[mcol];
    const float lwm = glw[mcol], lhm = glh[mcol];
#pragma unroll
    for (int g = 0; g < 8; ++g) {
      const int r = rsel + g;
      const float dx = fabsf(sRow[0][r] - cxm) / sRow[2][r];
      const float dy = fabsf(sRow[1][r] - cym) / sRow[3][r];
      float geo = g0 * __logf(fmaxf(dx, 1e-3f)) + g1 * __logf(fmaxf(dy, 1e-3f)) +
                  g2 * (lwm - sRow[4][r]) + g3 * (lhm - sRow[5][r]) + gB;
      geo = fmaxf(geo, 0.f);
      const float wt = fmaxf(geo * __expf(app[g] * scale), 1e-4f);
      z8[g] += wt;
      sW[r][mloc + cn] = (__bf16)wt;
    }
    __syncthreads();
#pragma unroll
    for (int kk = 0; kk < 64; kk += 32) {
      v16bf a = load_a_frag(&sW[0][kk], 64);
      accd0 = wmma_bf16(a, load_b_frag_T(&sVT[d0][kk], 64), accd0);
      accd1 = wmma_bf16(a, load_b_frag_T(&sVT[d0 + 16][kk], 64), accd1);
    }
  }
  __syncthreads();
#pragma unroll
  for (int g = 0; g < 8; ++g) atomicAdd(&sZ[rsel + g], z8[g]);
  __syncthreads();
  const float ws0 = Ws[d0 + cn], ws1 = Ws[d0 + 16 + cn];
#pragma unroll
  for (int g = 0; g < 8; ++g) {
    const float inv = 1.f / sZ[rsel + g];
    atomicAdd(&sLogit[rsel + g], (accd0[g] * ws0 + accd1[g] * ws1) * inv);
  }
  __syncthreads();
  if (tid < 16) {
    const float logit = sLogit[tid] + bs[0];
    out[rb + tid] = scores[rb + tid] * (1.f / (1.f + __expf(-logit)));
  }
}

// ---------------------------------------------------------------------------
extern "C" void kernel_launch(void* const* d_in, const int* in_sizes, int n_in,
                              void* d_out, int out_size, void* d_ws, size_t ws_size,
                              hipStream_t stream) {
  (void)in_sizes; (void)n_in; (void)out_size; (void)ws_size;
  const float* feature_map = (const float*)d_in[0];
  const float* boxes   = (const float*)d_in[1];
  const float* scores  = (const float*)d_in[2];
  const float* enc     = (const float*)d_in[3];
  const float* W_rank  = (const float*)d_in[4];
  const float* b_rank  = (const float*)d_in[5];
  const float* W_feat  = (const float*)d_in[6];
  const float* b_feat  = (const float*)d_in[7];
  const float* Wg1     = (const float*)d_in[8];
  const float* bg1     = (const float*)d_in[9];
  const float* Wg2     = (const float*)d_in[10];
  const float* bg2     = (const float*)d_in[11];
  const float* Wk      = (const float*)d_in[12];
  const float* bk      = (const float*)d_in[13];
  const float* Wq      = (const float*)d_in[14];
  const float* bq      = (const float*)d_in[15];
  const float* Wv      = (const float*)d_in[16];
  const float* bv      = (const float*)d_in[17];
  const float* Ws      = (const float*)d_in[18];
  const float* bs      = (const float*)d_in[19];

  char* ws = (char*)d_ws;
  int*    order  = (int*)(ws + OFF_ORDER);
  float*  gcx    = (float*)(ws + OFF_GEOM + 0 * 16384);
  float*  gcy    = (float*)(ws + OFF_GEOM + 1 * 16384);
  float*  gw     = (float*)(ws + OFF_GEOM + 2 * 16384);
  float*  gh     = (float*)(ws + OFF_GEOM + 3 * 16384);
  float*  glw    = (float*)(ws + OFF_GEOM + 4 * 16384);
  float*  glh    = (float*)(ws + OFF_GEOM + 5 * 16384);
  float*  wge    = (float*)(ws + OFF_WGE);
  __bf16* dWrank = (__bf16*)(ws + OFF_WRANK);
  __bf16* dWfeat = (__bf16*)(ws + OFF_WFEAT);
  __bf16* dWk    = (__bf16*)(ws + OFF_WK);
  __bf16* dWq    = (__bf16*)(ws + OFF_WQ);
  __bf16* dWv    = (__bf16*)(ws + OFF_WV);
  __bf16* kbuf   = (__bf16*)(ws + OFF_KB);
  __bf16* qbuf   = (__bf16*)(ws + OFF_QB);
  __bf16* vT     = (__bf16*)(ws + OFF_VT);

  prep_convert_kernel<<<384, 256, 0, stream>>>(
      W_rank, W_feat, Wk, Wq, Wv, Wg1, bg1, Wg2, bg2,
      dWrank, dWfeat, dWk, dWq, dWv, wge);

  prep_boxes_kernel<<<16, 256, 0, stream>>>(boxes, gcx, gcy, gw, gh, glw, glh);

  argsort_kernel<<<kB, kN, 0, stream>>>(scores, order);

  emb_kqv_kernel<<<(kB * kN) / 16, 128, 0, stream>>>(
      feature_map, enc, order, dWrank, dWfeat, b_rank, b_feat,
      dWk, dWq, dWv, bk, bq, bv, kbuf, qbuf, vT);

  relation_kernel<<<(kB * kN) / 16, 128, 0, stream>>>(
      kbuf, qbuf, vT, gcx, gcy, gw, gh, glw, glh, wge, Ws, bs,
      scores, (float*)d_out);
}
